// _GroupedMMExperts_5007931867137
// MI455X (gfx1250) — matrix operand order repaired and checked by reading, not compiled, not run on hardware
//
#include <hip/hip_runtime.h>

// ---------------------------------------------------------------------------
// Grouped MoE FFN: out = ragged_dot(ragged_dot(x, w1^T, gs), w2^T, gs)
// T=8192 tokens, E=8 experts, DIM=1024, HID=2048, bf16 math / f32 accum.
// Both GEMMs are "NT" (A and B K-contiguous in memory) -> no transposes.
// ---------------------------------------------------------------------------

#define T_TOK 8192
#define N_EXP 8
#define DIM_  1024
#define HID_  2048

#define BM 128
#define BN 128
#define BK 64
#define LDP 72   // LDS row pitch in bf16 elems (64 + 8 pad -> conflict-free b128 reads)

typedef __bf16 bf16_t;
typedef __attribute__((ext_vector_type(16))) __bf16 v16bf;
typedef __attribute__((ext_vector_type(8)))  __bf16 bf16x8;
typedef __attribute__((ext_vector_type(4)))  __bf16 bf16x4;
typedef __attribute__((ext_vector_type(8)))  float  v8f;
typedef __attribute__((ext_vector_type(4)))  float  f32x4;

// Native conversions: let the backend emit packed v_cvt bf16 ops instead of
// manual RNE bit-twiddling (which cost ~3 VALU ops per element).
__device__ __forceinline__ bf16_t cvt_bf16(float f) {
    return static_cast<bf16_t>(f);
}
__device__ __forceinline__ float bf16_to_f32(bf16_t b) {
    return static_cast<float>(b);
}
__device__ __forceinline__ bf16x4 cvt4(f32x4 f) {
    return __builtin_convertvector(f, bf16x4);
}

// A-fragment (16x32 bf16), ISA 7.12.2 layout:
//   lanes 0-15 : row M=lane, K = k0+0..7  and k0+16..23
//   lanes16-31 : row M=lane-16, K = k0+8..15 and k0+24..31
__device__ __forceinline__ v16bf load_frag_a(const bf16_t* tile, int k0, int lane) {
    const bf16_t* p = tile + (lane & 15) * LDP + k0 + ((lane >> 4) << 3);
    bf16x8 lo = *(const bf16x8*)p;
    bf16x8 hi = *(const bf16x8*)(p + 16);
    return __builtin_shufflevector(lo, hi, 0,1,2,3,4,5,6,7,8,9,10,11,12,13,14,15);
}
// B-fragment (32x16 bf16): lane n holds column n; lanes 0-15 K=k0+0..7 and k0+16..23,
// lanes 16-31 K=k0+8..15 and k0+24..31 (same order as A).
__device__ __forceinline__ v16bf load_frag_b(const bf16_t* tile, int k0, int lane) {
    const bf16_t* p = tile + (lane & 15) * LDP + k0 + ((lane >> 4) << 3);
    bf16x8 lo = *(const bf16x8*)p;
    bf16x8 hi = *(const bf16x8*)(p + 16);
    return __builtin_shufflevector(lo, hi, 0,1,2,3,4,5,6,7,8,9,10,11,12,13,14,15);
}

// One kernel for both layers.
//   A_F32   : A operand is fp32 in global (layer 1: x) vs bf16 (layer 2: h)
//   OUT_BF16: write bf16 (h) vs fp32 (final out, rounded through bf16)
//   KT      : reduction length, NT : output width; weights are [e][NT][KT].
template <bool A_F32, bool OUT_BF16, int KT, int NT>
__global__ __launch_bounds__(256) void moe_gemm_wmma(
    const void* __restrict__ Ap, const float* __restrict__ W,
    const int* __restrict__ counts, void* __restrict__ Outp) {

    __shared__ bf16_t As[BM * LDP];   // activations tile  [BM][BK] bf16
    __shared__ bf16_t Bs[BN * LDP];   // weight tile       [BN][BK] bf16

    const int tid  = threadIdx.x;
    const int lane = tid & 31;
    const int wv   = tid >> 5;          // 8 waves
    const int wm   = (wv & 1) << 6;     // wave M offset: 0 / 64
    const int wn   = (wv >> 1) << 5;    // wave N offset: 0/32/64/96
    const int m0   = blockIdx.y * BM;
    const int n0   = blockIdx.x * BN;

    // Expert for this M tile (tiles never straddle experts when counts are
    // multiples of BM, as in this pre-grouped configuration).
    int e = 0, seg = 0;
#pragma unroll
    for (int i = 0; i < N_EXP; ++i) {
        int c = counts[i];
        if (m0 >= seg + c) { seg += c; ++e; }
    }
    const float* We = W + (size_t)e * (size_t)NT * (size_t)KT;

    // Register staging buffers (double-buffered against LDS).
    f32x4  wr[8];
    f32x4  arf[8];
    bf16x8 arb[4];

    const int c4  = (tid & 15) << 2;   // f32 chunk col (4 floats)
    const int r0f = tid >> 4;          // 0..15
    const int c8  = (tid & 7) << 3;    // bf16 chunk col (8 bf16)
    const int r0b = tid >> 3;          // 0..31

    auto load_w = [&](int kb) {
#pragma unroll
        for (int i = 0; i < 8; ++i) {
            int r = r0f + (i << 4);
            wr[i] = *(const f32x4*)(We + (size_t)(n0 + r) * KT + kb + c4);
        }
    };
    auto load_a = [&](int kb) {
        if constexpr (A_F32) {
            const float* Af = (const float*)Ap;
#pragma unroll
            for (int i = 0; i < 8; ++i) {
                int r = r0f + (i << 4);
                arf[i] = *(const f32x4*)(Af + (size_t)(m0 + r) * KT + kb + c4);
            }
        } else {
            const bf16_t* Ab = (const bf16_t*)Ap;
#pragma unroll
            for (int i = 0; i < 4; ++i) {
                int r = r0b + (i << 5);
                arb[i] = *(const bf16x8*)(Ab + (size_t)(m0 + r) * KT + kb + c8);
            }
        }
    };
    auto store_lds = [&]() {
#pragma unroll
        for (int i = 0; i < 8; ++i) {
            int r = r0f + (i << 4);
            *(bf16x4*)&Bs[r * LDP + c4] = cvt4(wr[i]);
        }
        if constexpr (A_F32) {
#pragma unroll
            for (int i = 0; i < 8; ++i) {
                int r = r0f + (i << 4);
                *(bf16x4*)&As[r * LDP + c4] = cvt4(arf[i]);
            }
        } else {
#pragma unroll
            for (int i = 0; i < 4; ++i) {
                int r = r0b + (i << 5);
                *(bf16x8*)&As[r * LDP + c8] = arb[i];
            }
        }
    };

    v8f acc[4][2] = {};

    constexpr int NK = KT / BK;
    load_a(0);
    load_w(0);
#pragma unroll 1
    for (int kk = 0; kk < NK; ++kk) {
        __syncthreads();                 // previous compute done reading LDS
        store_lds();
        __syncthreads();
        if (kk + 1 < NK) {               // issue next global loads early:
            load_a((kk + 1) * BK);       // latency hides under WMMA below
            load_w((kk + 1) * BK);
        }
#pragma unroll
        for (int ks = 0; ks < BK; ks += 32) {
            v16bf b0 = load_frag_b(&Bs[(wn +  0) * LDP], ks, lane);
            v16bf b1 = load_frag_b(&Bs[(wn + 16) * LDP], ks, lane);
#pragma unroll
            for (int i = 0; i < 4; ++i) {
                v16bf a = load_frag_a(&As[(wm + (i << 4)) * LDP], ks, lane);
                acc[i][0] = __builtin_amdgcn_wmma_f32_16x16x32_bf16(
                    false, a, false, b0, (short)0, acc[i][0], false, false);
                acc[i][1] = __builtin_amdgcn_wmma_f32_16x16x32_bf16(
                    false, a, false, b1, (short)0, acc[i][1], false, false);
            }
        }
    }

    // Epilogue. C/D layout: VGPR v -> M = v + 8*(lane>=16), N = lane&15.
#pragma unroll
    for (int i = 0; i < 4; ++i) {
#pragma unroll
        for (int j = 0; j < 2; ++j) {
            int mb  = m0 + wm + (i << 4) + ((lane >> 4) << 3);
            int col = n0 + wn + (j << 4) + (lane & 15);
#pragma unroll
            for (int v = 0; v < 8; ++v) {
                float f = acc[i][j][v];
                if constexpr (OUT_BF16) {
                    ((bf16_t*)Outp)[(size_t)(mb + v) * NT + col] = cvt_bf16(f);
                } else {
                    // reference: bf16 result cast back to f32
                    ((float*)Outp)[(size_t)(mb + v) * NT + col] =
                        bf16_to_f32(cvt_bf16(f));
                }
            }
        }
    }
}

extern "C" void kernel_launch(void* const* d_in, const int* in_sizes, int n_in,
                              void* d_out, int out_size, void* d_ws, size_t ws_size,
                              hipStream_t stream) {
    const float* x   = (const float*)d_in[0];   // [T, DIM]      f32
    const float* w1  = (const float*)d_in[1];   // [E, HID, DIM] f32
    const float* w2  = (const float*)d_in[2];   // [E, DIM, HID] f32
    const int*   cnt = (const int*)d_in[3];     // [E]           i32

    bf16_t* h = (bf16_t*)d_ws;                  // [T, HID] bf16 (32 MB)

    dim3 blk(256);
    dim3 g1(HID_ / BN, T_TOK / BM);             // 16 x 64 tiles
    dim3 g2(DIM_ / BN, T_TOK / BM);             //  8 x 64 tiles

    // Layer 1: h = bf16( x @ w1^T ) per expert
    moe_gemm_wmma<true, true, DIM_, HID_><<<g1, blk, 0, stream>>>(x, w1, cnt, h);
    // Layer 2: out = f32( bf16( h @ w2^T ) ) per expert
    moe_gemm_wmma<false, false, HID_, DIM_><<<g2, blk, 0, stream>>>(h, w2, cnt, d_out);
}
